// AttentionLayer_35072702939687
// MI455X (gfx1250) — compile-verified, matrix operand order checked by reading, not run on hardware
//
#include <hip/hip_runtime.h>

// ---------------------------------------------------------------------------
// CDNA5 (gfx1250) attention block pipeline, wave32 + v_wmma_f32_16x16x32_bf16
// 2x2 register-blocked WMMA tiles, bf16 intermediates, LDS-tiled transposes,
// and an async-to-LDS double-buffered cooperative GEMM for the attention
// contraction (ASYNCcnt / s_wait_asynccnt path).
// ---------------------------------------------------------------------------

typedef __attribute__((ext_vector_type(16))) __bf16 v16bf;
typedef __attribute__((ext_vector_type(8)))  __bf16 v8bf;
typedef __attribute__((ext_vector_type(4)))  __bf16 v4bf;
typedef __attribute__((ext_vector_type(8)))  float  v8f;

#define B_   32
#define C_   512
#define HW_  1024
#define D_   64
#define OQKV 640   // 2*D + C

// ---- workspace layout (bytes, all 256-aligned) ----------------------------
#define OFF_BETA  ((size_t)0)            // beta^T bf16 [32][1024][1024] = 64MB (logits, softmaxed in place)
#define OFF_XR    ((size_t)67108864)     // xr f32 [32][512][1024] = 64MB
#define OFF_HT    ((size_t)134217728)    // h^T bf16 [32][1024][512] = 32MB (reused for z)
#define OFF_V     ((size_t)167772160)    // v bf16 [32][512][1024] = 32MB (reused for y1)
#define OFF_QT    ((size_t)201326592)    // q^T bf16 [32][1024][64] = 4MB
#define OFF_KT    ((size_t)205520896)    // k^T bf16 [32][1024][64] = 4MB
#define OFF_WQB   ((size_t)209715200)    // W_qkv bf16 (640x512)
#define OFF_W1B   ((size_t)210370560)    // W1 bf16 (512x512)
#define OFF_W2B   ((size_t)210894848)    // W2 bf16 (512x512)
#define OFF_MEAN1 ((size_t)211419136)
#define OFF_RSTD1 ((size_t)211421184)
#define OFF_MEAN2 ((size_t)211423232)
#define OFF_RSTD2 ((size_t)211425280)

// ---------------------------------------------------------------------------
// Fragment loader: ISA 7.12.2 layout for 16x32 bf16 A (and 32x16 B via its
// [N][K] transpose-equivalent). Row-major [row][K] bf16, ld multiple of 8,
// 16B-aligned rows. lane<16 -> K {k0..k0+7, k0+16..k0+23},
// lane>=16 -> K {k0+8..k0+15, k0+24..k0+31}.
// ---------------------------------------------------------------------------
__device__ __forceinline__ v16bf load_frag_bf16(const __bf16* __restrict__ base,
                                                int ld, int row0, int k0) {
  const int lane = threadIdx.x & 31;
  const int r    = row0 + (lane & 15);
  const int kk   = k0 + ((lane >> 4) << 3);
  const __bf16* p = base + (size_t)r * (size_t)ld + kk;
  v8bf lo = *reinterpret_cast<const v8bf*>(p);
  v8bf hi = *reinterpret_cast<const v8bf*>(p + 16);
  v16bf f;
#pragma unroll
  for (int i = 0; i < 8; ++i) { f[i] = lo[i]; f[i + 8] = hi[i]; }
  return f;
}

#define WMMA_BF16(A, Bf, Cacc) \
  __builtin_amdgcn_wmma_f32_16x16x32_bf16(false, (A), false, (Bf), (short)0, (Cacc), false, false)

// ---------------------------------------------------------------------------
// Async global -> LDS 16-byte copy (gfx1250). VDST VGPR carries the LDS byte
// address: low 32 bits of the flat shared-aperture pointer (ISA 10.2).
// Tracked by ASYNCcnt.
// ---------------------------------------------------------------------------
__device__ __forceinline__ void async_copy_b128(void* lds_ptr, const void* gptr) {
  unsigned l = (unsigned)(size_t)lds_ptr;
  asm volatile("global_load_async_to_lds_b128 %0, %1, off"
               :: "v"(l), "v"(gptr) : "memory");
}
__device__ __forceinline__ void wait_async_le3() {
  asm volatile("s_wait_asynccnt 0x3" ::: "memory");
}
__device__ __forceinline__ void wait_async_0() {
  asm volatile("s_wait_asynccnt 0x0" ::: "memory");
}

// ---------------------------------------------------------------------------
// BatchNorm stats: per-channel mean / rstd over (B, HW)
// ---------------------------------------------------------------------------
__global__ void bn_stats_kernel(const float* __restrict__ x,
                                float* __restrict__ mean,
                                float* __restrict__ rstd) {
  __shared__ float s_sum[256];
  __shared__ float s_sq[256];
  const int ch  = blockIdx.x;
  const int tid = threadIdx.x;
  float s = 0.f, sq = 0.f;
  for (int idx = tid; idx < B_ * HW_; idx += 256) {
    const int b  = idx >> 10;
    const int hw = idx & (HW_ - 1);
    const float v = x[(size_t)b * C_ * HW_ + (size_t)ch * HW_ + hw];
    s += v; sq += v * v;
  }
  s_sum[tid] = s; s_sq[tid] = sq;
  __syncthreads();
  for (int off = 128; off > 0; off >>= 1) {
    if (tid < off) { s_sum[tid] += s_sum[tid + off]; s_sq[tid] += s_sq[tid + off]; }
    __syncthreads();
  }
  if (tid == 0) {
    const float m   = s_sum[0] * (1.f / (B_ * HW_));
    const float var = s_sq[0] * (1.f / (B_ * HW_)) - m * m;
    mean[ch] = m;
    rstd[ch] = rsqrtf(var + 1e-5f);
  }
}

// ---------------------------------------------------------------------------
// Apply BN (+optional ReLU), write transposed bf16 [b][hw][c].
// 64x64 tile per block through LDS: coalesced float4 reads, 8B bf16 writes.
// ---------------------------------------------------------------------------
__global__ void bn_apply_transpose_kernel(const float* __restrict__ x,
                                          const float* __restrict__ mean,
                                          const float* __restrict__ rstd,
                                          const float* __restrict__ g,
                                          const float* __restrict__ bb,
                                          __bf16* __restrict__ out,
                                          int do_relu) {
  __shared__ __bf16 tile[64][68];     // [hw_local][c_local], padded
  const int b   = blockIdx.z;
  const int c0  = blockIdx.y * 64;
  const int hw0 = blockIdx.x * 64;
  const int tx  = threadIdx.x & 15;   // 16 x 16 thread layout
  const int ty  = threadIdx.x >> 4;
#pragma unroll
  for (int i = 0; i < 4; ++i) {
    const int c = c0 + ty + i * 16;
    const float gg  = g[c] * rstd[c];
    const float bbv = bb[c] - mean[c] * gg;
    const float4 v4 = *reinterpret_cast<const float4*>(
        &x[((size_t)b * C_ + c) * HW_ + hw0 + tx * 4]);
    float vals[4] = {v4.x, v4.y, v4.z, v4.w};
#pragma unroll
    for (int j = 0; j < 4; ++j) {
      float v = vals[j] * gg + bbv;
      if (do_relu) v = fmaxf(v, 0.f);
      tile[tx * 4 + j][ty + i * 16] = (__bf16)v;   // transpose into LDS
    }
  }
  __syncthreads();
#pragma unroll
  for (int i = 0; i < 4; ++i) {
    const int hwl = ty + i * 16;
    v4bf o;
#pragma unroll
    for (int j = 0; j < 4; ++j) o[j] = tile[hwl][tx * 4 + j];
    *reinterpret_cast<v4bf*>(
        &out[((size_t)b * HW_ + hw0 + hwl) * C_ + c0 + tx * 4]) = o;
  }
}

__global__ void f32_to_bf16_kernel(const float* __restrict__ in,
                                   __bf16* __restrict__ out, int n) {
  const int i = blockIdx.x * 256 + threadIdx.x;
  if (i < n) out[i] = (__bf16)in[i];
}

// ---------------------------------------------------------------------------
// GEMM1: qkv[b] = W_qkv (640x512) @ h[b] (512x1024) + b_qkv
// ---------------------------------------------------------------------------
__global__ void gemm_qkv_kernel(const __bf16* __restrict__ Wq,
                                const __bf16* __restrict__ ht,
                                const float* __restrict__ bqkv,
                                __bf16* __restrict__ qT,
                                __bf16* __restrict__ kT,
                                __bf16* __restrict__ vmat) {
  const int b  = blockIdx.z;
  const int t  = blockIdx.x * 8 + (threadIdx.x >> 5);   // 0..639 (20x32 tiles)
  const int m0 = (t >> 5) * 32;
  const int n0 = (t & 31) * 32;
  const __bf16* Bp = ht + (size_t)b * (HW_ * C_);
  v8f acc[2][2] = {};
  for (int k = 0; k < C_; k += 32) {
    const v16bf a0 = load_frag_bf16(Wq, C_, m0,      k);
    const v16bf a1 = load_frag_bf16(Wq, C_, m0 + 16, k);
    const v16bf b0 = load_frag_bf16(Bp, C_, n0,      k);
    const v16bf b1 = load_frag_bf16(Bp, C_, n0 + 16, k);
    acc[0][0] = WMMA_BF16(a0, b0, acc[0][0]);
    acc[0][1] = WMMA_BF16(a0, b1, acc[0][1]);
    acc[1][0] = WMMA_BF16(a1, b0, acc[1][0]);
    acc[1][1] = WMMA_BF16(a1, b1, acc[1][1]);
  }
  const int lane = threadIdx.x & 31;
  const int ncol = lane & 15;
  const int mrow = (lane >> 4) << 3;
#pragma unroll
  for (int si = 0; si < 2; ++si)
#pragma unroll
    for (int sj = 0; sj < 2; ++sj)
#pragma unroll
      for (int r = 0; r < 8; ++r) {
        const int m = m0 + si * 16 + mrow + r;
        const int n = n0 + sj * 16 + ncol;
        const float val = acc[si][sj][r] + bqkv[m];
        if (m < D_) {
          qT[(size_t)b * (HW_ * D_) + (size_t)n * D_ + m] = (__bf16)val;
        } else if (m < 2 * D_) {
          kT[(size_t)b * (HW_ * D_) + (size_t)n * D_ + (m - D_)] = (__bf16)val;
        } else {
          vmat[(size_t)b * (C_ * HW_) + (size_t)(m - 2 * D_) * HW_ + n] = (__bf16)val;
        }
      }
}

// ---------------------------------------------------------------------------
// GEMM2: beta^T[b][k][q] = (k^T . q^T) / sqrt(D). Logits written bf16.
// ---------------------------------------------------------------------------
__global__ void gemm_beta_kernel(const __bf16* __restrict__ kT,
                                 const __bf16* __restrict__ qT,
                                 __bf16* __restrict__ beta) {
  const int b  = blockIdx.z;
  const int t  = blockIdx.x * 8 + (threadIdx.x >> 5);   // 0..1023 (32x32 tiles)
  const int m0 = (t >> 5) * 32;
  const int n0 = (t & 31) * 32;
  const __bf16* Ap = kT + (size_t)b * (HW_ * D_);
  const __bf16* Bp = qT + (size_t)b * (HW_ * D_);
  v8f acc[2][2] = {};
#pragma unroll
  for (int k = 0; k < D_; k += 32) {
    const v16bf a0 = load_frag_bf16(Ap, D_, m0,      k);
    const v16bf a1 = load_frag_bf16(Ap, D_, m0 + 16, k);
    const v16bf b0 = load_frag_bf16(Bp, D_, n0,      k);
    const v16bf b1 = load_frag_bf16(Bp, D_, n0 + 16, k);
    acc[0][0] = WMMA_BF16(a0, b0, acc[0][0]);
    acc[0][1] = WMMA_BF16(a0, b1, acc[0][1]);
    acc[1][0] = WMMA_BF16(a1, b0, acc[1][0]);
    acc[1][1] = WMMA_BF16(a1, b1, acc[1][1]);
  }
  const int lane = threadIdx.x & 31;
  const int ncol = lane & 15;
  const int mrow = (lane >> 4) << 3;
  __bf16* out = beta + (size_t)b * ((size_t)HW_ * HW_);
#pragma unroll
  for (int si = 0; si < 2; ++si)
#pragma unroll
    for (int sj = 0; sj < 2; ++sj)
#pragma unroll
      for (int r = 0; r < 8; ++r) {
        const int m = m0 + si * 16 + mrow + r;
        const int n = n0 + sj * 16 + ncol;
        out[(size_t)m * HW_ + n] = (__bf16)(acc[si][sj][r] * 0.125f);
      }
}

// ---------------------------------------------------------------------------
// In-place row softmax over beta^T rows (softmax over q for fixed k).
// ---------------------------------------------------------------------------
__global__ void softmax_rows_kernel(__bf16* __restrict__ beta) {
  __shared__ float red[256];
  const int tid = threadIdx.x;
  __bf16* p = beta + (size_t)blockIdx.x * HW_;
  float v[4];
#pragma unroll
  for (int j = 0; j < 4; ++j) v[j] = (float)p[tid + j * 256];
  float mx = fmaxf(fmaxf(v[0], v[1]), fmaxf(v[2], v[3]));
  red[tid] = mx;
  __syncthreads();
  for (int off = 128; off > 0; off >>= 1) {
    if (tid < off) red[tid] = fmaxf(red[tid], red[tid + off]);
    __syncthreads();
  }
  mx = red[0];
  __syncthreads();
  float e[4];
  float s = 0.f;
#pragma unroll
  for (int j = 0; j < 4; ++j) { e[j] = __expf(v[j] - mx); s += e[j]; }
  red[tid] = s;
  __syncthreads();
  for (int off = 128; off > 0; off >>= 1) {
    if (tid < off) red[tid] += red[tid + off];
    __syncthreads();
  }
  const float inv = 1.f / red[0];
#pragma unroll
  for (int j = 0; j < 4; ++j) p[tid + j * 256] = (__bf16)(e[j] * inv);
}

// ---------------------------------------------------------------------------
// GEMM3 (async-LDS): att[b] = v (512x1024) @ beta (1024x1024); xr = x + att.
// Block tile 64x128 (8 waves as 2x4, each wave 2x2 WMMA tiles of 16x16).
// K panels (A 64x32, B 128x32) staged with global_load_async_to_lds_b128,
// double-buffered; fragments then read from LDS (ds path).
// ---------------------------------------------------------------------------
#define LDPAD 48   // padded row length (elements) for LDS panels, 16B aligned
__global__ void gemm_att_kernel(const __bf16* __restrict__ vmat,
                                const __bf16* __restrict__ beta,
                                const float* __restrict__ x,
                                float* __restrict__ xr) {
  __shared__ __bf16 Abuf[2][64][LDPAD];    // 12 KB
  __shared__ __bf16 Bbuf[2][128][LDPAD];   // 24 KB
  const int b    = blockIdx.z;
  const int m0   = (blockIdx.x >> 3) * 64;    // c block
  const int n0   = (blockIdx.x & 7) * 128;    // k_hw block
  const int tid  = threadIdx.x;
  const int wave = tid >> 5;
  const int wm   = wave >> 2;                 // 0..1
  const int wn   = wave & 3;                  // 0..3
  const __bf16* Ap = vmat + (size_t)b * (C_ * HW_);
  const __bf16* Bp = beta + (size_t)b * ((size_t)HW_ * HW_);

  // panel-copy assignments (per thread): 1x A b128 + 2x B b128
  const int ar = tid >> 2, as = (tid & 3) * 8;          // A: 64 rows x 4 segs
  const int br0 = tid >> 1, bs0 = (tid & 1) * 16;       // B: ids 2*tid, 2*tid+1
  // id = 2*tid+j -> row = id>>2, seg = (id&3)*8 ; with j in {0,1} this is
  // row = tid>>1, segs {(tid&1)*16, (tid&1)*16+8}

  // preload k=0 into buffer 0
  async_copy_b128(&Abuf[0][ar][as], Ap + (size_t)(m0 + ar) * HW_ + as);
  async_copy_b128(&Bbuf[0][br0][bs0], Bp + (size_t)(n0 + br0) * HW_ + bs0);
  async_copy_b128(&Bbuf[0][br0][bs0 + 8], Bp + (size_t)(n0 + br0) * HW_ + bs0 + 8);

  v8f acc[2][2] = {};
  const int lane = tid & 31;
  const int rsel = lane & 15;
  const int ksel = (lane >> 4) << 3;

  int buf = 0;
  for (int kb = 0; kb < HW_ / 32; ++kb) {
    if (kb + 1 < HW_ / 32) {
      const int k1 = (kb + 1) * 32;
      async_copy_b128(&Abuf[buf ^ 1][ar][as], Ap + (size_t)(m0 + ar) * HW_ + k1 + as);
      async_copy_b128(&Bbuf[buf ^ 1][br0][bs0], Bp + (size_t)(n0 + br0) * HW_ + k1 + bs0);
      async_copy_b128(&Bbuf[buf ^ 1][br0][bs0 + 8], Bp + (size_t)(n0 + br0) * HW_ + k1 + bs0 + 8);
      wait_async_le3();   // current panel complete; next 3 still in flight
    } else {
      wait_async_0();
    }
    __syncthreads();      // all waves' panel writes visible

    v16bf af[2], bfr[2];
#pragma unroll
    for (int si = 0; si < 2; ++si) {
      const __bf16* pa = &Abuf[buf][wm * 32 + si * 16 + rsel][ksel];
      v8bf lo = *reinterpret_cast<const v8bf*>(pa);
      v8bf hi = *reinterpret_cast<const v8bf*>(pa + 16);
#pragma unroll
      for (int i = 0; i < 8; ++i) { af[si][i] = lo[i]; af[si][i + 8] = hi[i]; }
    }
#pragma unroll
    for (int sj = 0; sj < 2; ++sj) {
      const __bf16* pb = &Bbuf[buf][wn * 32 + sj * 16 + rsel][ksel];
      v8bf lo = *reinterpret_cast<const v8bf*>(pb);
      v8bf hi = *reinterpret_cast<const v8bf*>(pb + 16);
#pragma unroll
      for (int i = 0; i < 8; ++i) { bfr[sj][i] = lo[i]; bfr[sj][i + 8] = hi[i]; }
    }
    acc[0][0] = WMMA_BF16(af[0], bfr[0], acc[0][0]);
    acc[0][1] = WMMA_BF16(af[0], bfr[1], acc[0][1]);
    acc[1][0] = WMMA_BF16(af[1], bfr[0], acc[1][0]);
    acc[1][1] = WMMA_BF16(af[1], bfr[1], acc[1][1]);

    __syncthreads();      // reads done before this buffer is refilled
    buf ^= 1;
  }

  const int ncol = lane & 15;
  const int mrow = (lane >> 4) << 3;
#pragma unroll
  for (int si = 0; si < 2; ++si)
#pragma unroll
    for (int sj = 0; sj < 2; ++sj)
#pragma unroll
      for (int r = 0; r < 8; ++r) {
        const int m = m0 + wm * 32 + si * 16 + mrow + r;
        const int n = n0 + wn * 32 + sj * 16 + ncol;
        const size_t idx = (size_t)b * (C_ * HW_) + (size_t)m * HW_ + n;
        xr[idx] = x[idx] + acc[si][sj][r];
      }
}

// ---------------------------------------------------------------------------
// GEMM4: y1[b][hw][j] = relu(z[b][hw][:] @ W1[j][:] + b1[j])   (bf16 out)
// ---------------------------------------------------------------------------
__global__ void gemm_mlp1_kernel(const __bf16* __restrict__ z,
                                 const __bf16* __restrict__ W1b,
                                 const float* __restrict__ b1,
                                 __bf16* __restrict__ y1) {
  const int b  = blockIdx.z;
  const int t  = blockIdx.x * 8 + (threadIdx.x >> 5);   // 0..511 (32x16 tiles)
  const int m0 = (t >> 4) * 32;                         // hw
  const int n0 = (t & 15) * 32;                         // j
  const __bf16* Ap = z + (size_t)b * (HW_ * C_);
  v8f acc[2][2] = {};
  for (int k = 0; k < C_; k += 32) {
    const v16bf a0 = load_frag_bf16(Ap, C_, m0,      k);
    const v16bf a1 = load_frag_bf16(Ap, C_, m0 + 16, k);
    const v16bf b0 = load_frag_bf16(W1b, C_, n0,      k);
    const v16bf b1 = load_frag_bf16(W1b, C_, n0 + 16, k);
    acc[0][0] = WMMA_BF16(a0, b0, acc[0][0]);
    acc[0][1] = WMMA_BF16(a0, b1, acc[0][1]);
    acc[1][0] = WMMA_BF16(a1, b0, acc[1][0]);
    acc[1][1] = WMMA_BF16(a1, b1, acc[1][1]);
  }
  const int lane = threadIdx.x & 31;
  const int ncol = lane & 15;
  const int mrow = (lane >> 4) << 3;
#pragma unroll
  for (int si = 0; si < 2; ++si)
#pragma unroll
    for (int sj = 0; sj < 2; ++sj) {
      const int n = n0 + sj * 16 + ncol;
      const float bias = b1[n];
#pragma unroll
      for (int r = 0; r < 8; ++r) {
        const int m = m0 + si * 16 + mrow + r;
        const float val = fmaxf(acc[si][sj][r] + bias, 0.f);
        y1[(size_t)b * (HW_ * C_) + (size_t)m * C_ + n] = (__bf16)val;
      }
    }
}

// ---------------------------------------------------------------------------
// GEMM5: out[b][i][hw] = xr[b][i][hw] + y1[b][hw][:] @ W2[i][:] + b2[i]
// ---------------------------------------------------------------------------
__global__ void gemm_mlp2_kernel(const __bf16* __restrict__ y1,
                                 const __bf16* __restrict__ W2b,
                                 const float* __restrict__ b2,
                                 const float* __restrict__ xr,
                                 float* __restrict__ out) {
  const int b  = blockIdx.z;
  const int t  = blockIdx.x * 8 + (threadIdx.x >> 5);   // 0..511 (32x16 tiles)
  const int m0 = (t >> 4) * 32;                         // hw
  const int n0 = (t & 15) * 32;                         // i
  const __bf16* Ap = y1 + (size_t)b * (HW_ * C_);
  v8f acc[2][2] = {};
  for (int k = 0; k < C_; k += 32) {
    const v16bf a0 = load_frag_bf16(Ap, C_, m0,      k);
    const v16bf a1 = load_frag_bf16(Ap, C_, m0 + 16, k);
    const v16bf b0 = load_frag_bf16(W2b, C_, n0,      k);
    const v16bf b1 = load_frag_bf16(W2b, C_, n0 + 16, k);
    acc[0][0] = WMMA_BF16(a0, b0, acc[0][0]);
    acc[0][1] = WMMA_BF16(a0, b1, acc[0][1]);
    acc[1][0] = WMMA_BF16(a1, b0, acc[1][0]);
    acc[1][1] = WMMA_BF16(a1, b1, acc[1][1]);
  }
  const int lane = threadIdx.x & 31;
  const int ncol = lane & 15;
  const int mrow = (lane >> 4) << 3;
#pragma unroll
  for (int si = 0; si < 2; ++si)
#pragma unroll
    for (int sj = 0; sj < 2; ++sj) {
      const int n = n0 + sj * 16 + ncol;
      const float bias = b2[n];
#pragma unroll
      for (int r = 0; r < 8; ++r) {
        const int m = m0 + si * 16 + mrow + r;
        const size_t idx = (size_t)b * (C_ * HW_) + (size_t)n * HW_ + m;
        out[idx] = xr[idx] + acc[si][sj][r] + bias;
      }
    }
}

// ---------------------------------------------------------------------------
extern "C" void kernel_launch(void* const* d_in, const int* in_sizes, int n_in,
                              void* d_out, int out_size, void* d_ws, size_t ws_size,
                              hipStream_t stream) {
  (void)in_sizes; (void)n_in; (void)out_size; (void)ws_size;
  const float* x     = (const float*)d_in[0];
  const float* bn1_g = (const float*)d_in[1];
  const float* bn1_b = (const float*)d_in[2];
  const float* W_qkv = (const float*)d_in[3];
  const float* b_qkv = (const float*)d_in[4];
  const float* bn2_g = (const float*)d_in[5];
  const float* bn2_b = (const float*)d_in[6];
  const float* W1    = (const float*)d_in[7];
  const float* b1    = (const float*)d_in[8];
  const float* W2    = (const float*)d_in[9];
  const float* b2    = (const float*)d_in[10];
  float* out = (float*)d_out;

  char* ws = (char*)d_ws;
  __bf16* beta  = (__bf16*)(ws + OFF_BETA);
  float*  xr    = (float*)(ws + OFF_XR);
  __bf16* ht    = (__bf16*)(ws + OFF_HT);    // reused as z
  __bf16* vmat  = (__bf16*)(ws + OFF_V);     // reused as y1
  __bf16* qT    = (__bf16*)(ws + OFF_QT);
  __bf16* kT    = (__bf16*)(ws + OFF_KT);
  __bf16* WqB   = (__bf16*)(ws + OFF_WQB);
  __bf16* W1B   = (__bf16*)(ws + OFF_W1B);
  __bf16* W2B   = (__bf16*)(ws + OFF_W2B);
  float*  mean1 = (float*)(ws + OFF_MEAN1);
  float*  rstd1 = (float*)(ws + OFF_RSTD1);
  float*  mean2 = (float*)(ws + OFF_MEAN2);
  float*  rstd2 = (float*)(ws + OFF_RSTD2);

  // 1) BN1 stats
  bn_stats_kernel<<<C_, 256, 0, stream>>>(x, mean1, rstd1);
  // 2) BN1 + ReLU -> h^T bf16 [b][hw][c]  (LDS-tiled transpose)
  bn_apply_transpose_kernel<<<dim3(HW_ / 64, C_ / 64, B_), 256, 0, stream>>>(
      x, mean1, rstd1, bn1_g, bn1_b, ht, 1);
  // 3) weights -> bf16
  f32_to_bf16_kernel<<<(OQKV * C_ + 255) / 256, 256, 0, stream>>>(W_qkv, WqB, OQKV * C_);
  f32_to_bf16_kernel<<<(C_ * C_ + 255) / 256, 256, 0, stream>>>(W1, W1B, C_ * C_);
  f32_to_bf16_kernel<<<(C_ * C_ + 255) / 256, 256, 0, stream>>>(W2, W2B, C_ * C_);
  // 4) qkv GEMM  (20x32 wave tiles of 32x32, 8 waves/block)
  gemm_qkv_kernel<<<dim3(80, 1, B_), 256, 0, stream>>>(WqB, ht, b_qkv, qT, kT, vmat);
  // 5) beta^T logits GEMM (32x32 wave tiles) -> bf16
  gemm_beta_kernel<<<dim3(128, 1, B_), 256, 0, stream>>>(kT, qT, beta);
  // 6) in-place softmax over q (contiguous rows of beta^T)
  softmax_rows_kernel<<<B_ * HW_, 256, 0, stream>>>(beta);
  // 7) att GEMM + residual -> xr  (async-LDS double-buffered, 8x8 block tiles)
  gemm_att_kernel<<<dim3(64, 1, B_), 256, 0, stream>>>(vmat, beta, x, xr);
  // 8) BN2 stats
  bn_stats_kernel<<<C_, 256, 0, stream>>>(xr, mean2, rstd2);
  // 9) BN2 (no relu) -> z bf16 [b][hw][c]  (reuses ht region)
  bn_apply_transpose_kernel<<<dim3(HW_ / 64, C_ / 64, B_), 256, 0, stream>>>(
      xr, mean2, rstd2, bn2_g, bn2_b, ht, 0);
  // 10) MLP1: relu(z @ W1^T + b1) -> y1 bf16 (reuses v region)
  gemm_mlp1_kernel<<<dim3(64, 1, B_), 256, 0, stream>>>(ht, W1B, b1, vmat);
  // 11) MLP2 + bias + residual -> out
  gemm_mlp2_kernel<<<dim3(64, 1, B_), 256, 0, stream>>>(vmat, W2B, b2, xr, out);
}
